// FCGF_point_att_k_final_89575837925682
// MI455X (gfx1250) — matrix operand order repaired and checked by reading, not compiled, not run on hardware
//
#include <hip/hip_runtime.h>
#include <math.h>

// ---------------------------------------------------------------------------
// FCGF point-attention pipeline for MI455X (gfx1250), wave32 + bf16 WMMA
// + async global->LDS staging (ASYNCcnt) where tiles are byte-copies.
// Assumptions fixed by setup_inputs(): B=16, L=8192, C=32 (N=131072).
// Biases feeding a BatchNorm are mathematically absorbed and skipped.
// ---------------------------------------------------------------------------

typedef __bf16 bf16;
typedef __attribute__((ext_vector_type(16))) __bf16 v16bf;
typedef __attribute__((ext_vector_type(8)))  float  v8f;
typedef __attribute__((ext_vector_type(4)))  int    v4i;

#define DEVI __device__ __forceinline__

#if defined(__has_builtin)
#  if __has_builtin(__builtin_amdgcn_global_load_async_to_lds_b128) && \
      __has_builtin(__builtin_amdgcn_s_wait_asynccnt)
#    define HAS_ASYNC_LDS 1
#  endif
#endif
#ifndef HAS_ASYNC_LDS
#  define HAS_ASYNC_LDS 0
#endif

#if HAS_ASYNC_LDS
typedef __attribute__((address_space(1))) v4i gas_v4i;  // global
typedef __attribute__((address_space(3))) v4i las_v4i;  // LDS
#endif

// copy 16 bytes global -> LDS via the async path (per-lane, ASYNCcnt tracked)
DEVI void async_copy16(const void* g, void* l) {
#if HAS_ASYNC_LDS
  __builtin_amdgcn_global_load_async_to_lds_b128((gas_v4i*)g, (las_v4i*)l, 0, 0);
#endif
  (void)g; (void)l;
}
DEVI void async_wait0() {
#if HAS_ASYNC_LDS
  __builtin_amdgcn_s_wait_asynccnt(0);
#endif
}

// ---- WMMA fragment gathers (wave32; contiguous k order, same in A and B) ----------
// A (16x32 bf16, MxK): lane l -> m=l&15, half=l>>4; elem e:
//   k = half*16 + e
//   => one contiguous 16-element run per lane: [m*32+half*16)
DEVI v16bf load_a_frag(const bf16* A, int ld, int lane) {
  int m = lane & 15, half = lane >> 4;
  v16bf a;
#pragma unroll
  for (int e = 0; e < 16; ++e) {
    int k = half * 16 + e;
    a[e] = A[m * ld + k];
  }
  return a;
}
// B (32x16 bf16, KxN) from an LDS tile stored [K][N]: elem e -> k = half*16+e
DEVI v16bf load_b_frag_kn(const bf16* Bp, int ld, int lane) {
  int n = lane & 15, half = lane >> 4;
  v16bf b;
#pragma unroll
  for (int e = 0; e < 16; ++e) b[e] = Bp[(half * 16 + e) * ld + n];
  return b;
}
// B fragment from an LDS tile stored row-major [N][32] (memory order of
// [oc][k] weights): per lane one contiguous 16-element (32B) run.
DEVI v16bf load_b_frag_rm(const bf16* Bp, int lane) {
  int n = lane & 15, half = lane >> 4;
  v16bf b;
#pragma unroll
  for (int e = 0; e < 16; ++e) b[e] = Bp[n * 32 + half * 16 + e];
  return b;
}
// C/D (16x16 f32): elem v, lane l -> row = v + 8*(l>>4), col = l&15

DEVI v8f wmma_bf16(v16bf a, v16bf b, v8f c) {
  return __builtin_amdgcn_wmma_f32_16x16x32_bf16(
      false, a, false, b, (short)0, c, false, false);
}

// ---------------------------------------------------------------------------
__global__ void zero_kernel(float* p, int n) {
  int i = blockIdx.x * blockDim.x + threadIdx.x;
  if (i < n) p[i] = 0.f;
}

__global__ void f2bf_kernel(const float* src, bf16* dst, long long n) {
  long long stride = (long long)gridDim.x * blockDim.x;
  for (long long i = (long long)blockIdx.x * blockDim.x + threadIdx.x; i < n; i += stride)
    dst[i] = (bf16)src[i];
}

// point-MLP layer1: h1 = x @ po_w1^T (bias absorbed by BN); also BN sums and
// per-segment channel sums of x (for channel attention).
__global__ void po1_kernel(const float* __restrict__ x, const float* __restrict__ w,
                           float* __restrict__ h1, float* s_po1, float* csum, int L) {
  __shared__ float ws[16 * 32];
  __shared__ float s1[16], s2[16], cz[32];
  int tid = threadIdx.x;
  for (int i = tid; i < 512; i += 256) ws[i] = w[i];
  if (tid < 16) { s1[tid] = 0.f; s2[tid] = 0.f; }
  if (tid < 32) cz[tid] = 0.f;
  __syncthreads();
  size_t n = (size_t)blockIdx.x * 256 + tid;
  float xv[32];
  const float4* xr = reinterpret_cast<const float4*>(x + n * 32);
#pragma unroll
  for (int q = 0; q < 8; ++q) {
    float4 t = xr[q];
    xv[4 * q] = t.x; xv[4 * q + 1] = t.y; xv[4 * q + 2] = t.z; xv[4 * q + 3] = t.w;
  }
  float acc[16];
#pragma unroll
  for (int j = 0; j < 16; ++j) {
    float a = 0.f;
#pragma unroll
    for (int i = 0; i < 32; ++i) a += xv[i] * ws[j * 32 + i];
    acc[j] = a;
  }
  float4* hr = reinterpret_cast<float4*>(h1 + n * 16);
#pragma unroll
  for (int q = 0; q < 4; ++q)
    hr[q] = make_float4(acc[4 * q], acc[4 * q + 1], acc[4 * q + 2], acc[4 * q + 3]);
#pragma unroll
  for (int j = 0; j < 16; ++j) { atomicAdd(&s1[j], acc[j]); atomicAdd(&s2[j], acc[j] * acc[j]); }
#pragma unroll
  for (int i = 0; i < 32; ++i) atomicAdd(&cz[i], xv[i]);
  __syncthreads();
  if (tid < 16) { atomicAdd(&s_po1[tid], s1[tid]); atomicAdd(&s_po1[16 + tid], s2[tid]); }
  if (tid < 32) {
    int b = (int)(((size_t)blockIdx.x * 256) / (size_t)L);  // 256 | L
    atomicAdd(&csum[b * 32 + tid], cz[tid]);
  }
}

// convert BN sums -> (scale, shift): bn(v) = scale*v + shift
__global__ void bn_finalize(const float* sums, const float* g, const float* be,
                            float* out, int Cc, float invN) {
  int t = blockIdx.x * blockDim.x + threadIdx.x;
  if (t < Cc) {
    float mu = sums[t] * invN;
    float var = sums[Cc + t] * invN - mu * mu;
    float sc = g[t] * rsqrtf(var + 1e-5f);
    out[t] = sc;
    out[Cc + t] = be[t] - mu * sc;
  }
}

// point-MLP layer2: z = relu(bn(h1)) @ po_w2^T  (scalar per point) + BN sums
__global__ void po2_kernel(const float* __restrict__ h1, const float* sc_po1,
                           const float* __restrict__ w2, float* z, float* s_po2) {
  __shared__ float red1[256], red2[256];
  int tid = threadIdx.x;
  size_t n = (size_t)blockIdx.x * 256 + tid;
  float a = 0.f;
#pragma unroll
  for (int j = 0; j < 16; ++j) {
    float v = fmaxf(sc_po1[j] * h1[n * 16 + j] + sc_po1[16 + j], 0.f);
    a += v * w2[j];
  }
  z[n] = a;
  red1[tid] = a; red2[tid] = a * a;
  __syncthreads();
  for (int s = 128; s > 0; s >>= 1) {
    if (tid < s) { red1[tid] += red1[tid + s]; red2[tid] += red2[tid + s]; }
    __syncthreads();
  }
  if (tid == 0) { atomicAdd(&s_po2[0], red1[0]); atomicAdd(&s_po2[1], red2[0]); }
}

// per-segment softmax of p_att = relu(bn(z))   (L <= 8192)
__global__ void pseg_kernel(const float* __restrict__ z, const float* sc_po2,
                            float* p_soft, int L) {
  __shared__ float pv[8192];
  __shared__ float red[256];
  int b = blockIdx.x, tid = threadIdx.x;
  float scale = sc_po2[0], shift = sc_po2[1];
  const float* zb = z + (size_t)b * L;
  float m = -1e30f;
  for (int i = tid; i < L; i += 256) {
    float p = fmaxf(scale * zb[i] + shift, 0.f);
    pv[i] = p; m = fmaxf(m, p);
  }
  red[tid] = m; __syncthreads();
  for (int s = 128; s > 0; s >>= 1) { if (tid < s) red[tid] = fmaxf(red[tid], red[tid + s]); __syncthreads(); }
  m = red[0]; __syncthreads();
  float s = 0.f;
  for (int i = tid; i < L; i += 256) { float e = __expf(pv[i] - m); pv[i] = e; s += e; }
  red[tid] = s; __syncthreads();
  for (int ss = 128; ss > 0; ss >>= 1) { if (tid < ss) red[tid] += red[tid + ss]; __syncthreads(); }
  float inv = 1.f / red[0];
  for (int i = tid; i < L; i += 256) p_soft[(size_t)b * L + i] = pv[i] * inv;
}

// channel attention MLP (tiny): one wave
__global__ void chmlp_kernel(const float* csum, const float* w1, const float* b1,
                             const float* w2, const float* b2, float* c_att,
                             int B, float invL) {
  __shared__ float c[32], h[16], o[32];
  __shared__ float mx, sm;
  int t = threadIdx.x;  // 32 threads
  for (int b = 0; b < B; ++b) {
    c[t] = csum[b * 32 + t] * invL;
    __syncthreads();
    if (t < 16) {
      float a = b1[t];
      for (int i = 0; i < 32; ++i) a += c[i] * w1[t * 32 + i];
      h[t] = fmaxf(a, 0.f);
    }
    __syncthreads();
    {
      float a = b2[t];
      for (int j = 0; j < 16; ++j) a += h[j] * w2[t * 16 + j];
      o[t] = fmaxf(a, 0.f);
    }
    __syncthreads();
    if (t == 0) {
      float m = o[0];
      for (int i = 1; i < 32; ++i) m = fmaxf(m, o[i]);
      float s = 0.f;
      for (int i = 0; i < 32; ++i) s += __expf(o[i] - m);
      mx = m; sm = s;
    }
    __syncthreads();
    c_att[b * 32 + t] = __expf(o[t] - mx) / sm;
    __syncthreads();
  }
}

// flt = x * p_soft * c_att * 1e7  -> bf16
__global__ void flt_kernel(const float* __restrict__ x, const float* __restrict__ p_soft,
                           const float* __restrict__ c_att, bf16* flt, int L) {
  size_t idx = (size_t)blockIdx.x * 256 + threadIdx.x;
  size_t n = idx >> 5;
  int cc = (int)(idx & 31);
  int b = (int)(n / (size_t)L);
  flt[idx] = (bf16)(x[idx] * p_soft[n] * c_att[b * 32 + cc] * 1e7f);
}

// GEMM1: k1_pre[N,128] = flt[N,32] @ k_w1^T   (K=32, one wmma per tile)
__global__ void gemm1_kernel(const bf16* __restrict__ A, const bf16* __restrict__ w1b,
                             float* __restrict__ Cout) {
  __shared__ __align__(16) bf16 Abuf[32 * 32];
  __shared__ __align__(16) bf16 Bbuf[64 * 32];  // row-major [oc][k]
  int tid = threadIdx.x, lane = tid & 31, wave = tid >> 5;
  size_t r0 = (size_t)blockIdx.x * 32;
  int oc0 = blockIdx.y * 64;
#if HAS_ASYNC_LDS
  // A tile: 32 rows x 32 bf16 = 2KB contiguous. B tile: 64 rows x 64B = 4KB contiguous.
  if (tid < 128) async_copy16(A + r0 * 32 + (size_t)tid * 8, (char*)Abuf + tid * 16);
  async_copy16(w1b + (size_t)oc0 * 32 + (size_t)tid * 8, (char*)Bbuf + tid * 16);
  async_wait0();
#else
#pragma unroll
  for (int q = 0; q < 4; ++q) {
    int idx = tid + q * 256; int r = idx >> 5, k = idx & 31;
    Abuf[r * 32 + k] = A[(r0 + r) * 32 + k];
  }
#pragma unroll
  for (int q = 0; q < 8; ++q) {
    int idx = tid + q * 256; int j = idx >> 5, k = idx & 31;
    Bbuf[j * 32 + k] = w1b[(oc0 + j) * 32 + k];
  }
#endif
  __syncthreads();
  int wtR = wave >> 2, wtC = wave & 3;
  v16bf a = load_a_frag(Abuf + wtR * 16 * 32, 32, lane);
  v16bf bb = load_b_frag_rm(Bbuf + wtC * 16 * 32, lane);
  v8f acc = {};
  acc = wmma_bf16(a, bb, acc);
  int m = lane & 15, half = lane >> 4;
#pragma unroll
  for (int v = 0; v < 8; ++v)
    Cout[(r0 + wtR * 16 + v + 8 * half) * 128 + oc0 + wtC * 16 + m] = acc[v];
}

__global__ void stats_f32_kernel(const float* __restrict__ src, float* sums,
                                 long long n_elems, int Cc) {
  __shared__ float ss[512], sq[512];
  for (int i = threadIdx.x; i < Cc; i += blockDim.x) { ss[i] = 0.f; sq[i] = 0.f; }
  __syncthreads();
  long long stride = (long long)gridDim.x * blockDim.x;
  for (long long i = (long long)blockIdx.x * blockDim.x + threadIdx.x; i < n_elems; i += stride) {
    float v = src[i]; int c = (int)(i & (long long)(Cc - 1));
    atomicAdd(&ss[c], v); atomicAdd(&sq[c], v * v);
  }
  __syncthreads();
  for (int i = threadIdx.x; i < Cc; i += blockDim.x) {
    atomicAdd(&sums[i], ss[i]); atomicAdd(&sums[Cc + i], sq[i]);
  }
}

__global__ void stats_bf16_kernel(const bf16* __restrict__ src, float* sums,
                                  long long n_elems, int Cc) {
  __shared__ float ss[512], sq[512];
  for (int i = threadIdx.x; i < Cc; i += blockDim.x) { ss[i] = 0.f; sq[i] = 0.f; }
  __syncthreads();
  long long stride = (long long)gridDim.x * blockDim.x;
  for (long long i = (long long)blockIdx.x * blockDim.x + threadIdx.x; i < n_elems; i += stride) {
    float v = (float)src[i]; int c = (int)(i & (long long)(Cc - 1));
    atomicAdd(&ss[c], v); atomicAdd(&sq[c], v * v);
  }
  __syncthreads();
  for (int i = threadIdx.x; i < Cc; i += blockDim.x) {
    atomicAdd(&sums[i], ss[i]); atomicAdd(&sums[Cc + i], sq[i]);
  }
}

// GEMM2: k2[N,512] = relu(bn(k1_pre)) @ k_w2^T   (K=128; BN+relu fused into A staging)
__global__ void gemm2_kernel(const float* __restrict__ k1, const bf16* __restrict__ w2b,
                             const float* sc_k1, bf16* __restrict__ k2out) {
  __shared__ __align__(16) bf16 Abuf[32 * 32];
  __shared__ __align__(16) bf16 Bbuf[64 * 32];  // row-major [oc][k-chunk]
  __shared__ float ssc[128], ssh[128];
  int tid = threadIdx.x, lane = tid & 31, wave = tid >> 5;
  size_t r0 = (size_t)blockIdx.x * 32;
  int oc0 = blockIdx.y * 64;
  if (tid < 128) { ssc[tid] = sc_k1[tid]; ssh[tid] = sc_k1[128 + tid]; }
  __syncthreads();
  int wtR = wave >> 2, wtC = wave & 3;
  v8f acc = {};
  for (int kk = 0; kk < 4; ++kk) {
#if HAS_ASYNC_LDS
    {  // 64 rows x 32 bf16; each thread moves one 16B quarter-row (overlaps A transform)
      int row = tid >> 2, c4 = tid & 3;
      async_copy16(w2b + (size_t)(oc0 + row) * 128 + kk * 32 + c4 * 8,
                   (char*)Bbuf + (row * 32 + c4 * 8) * 2);
    }
#else
#pragma unroll
    for (int q = 0; q < 8; ++q) {
      int idx = tid + q * 256; int j = idx >> 5, k = idx & 31;
      Bbuf[j * 32 + k] = w2b[(oc0 + j) * 128 + kk * 32 + k];
    }
#endif
#pragma unroll
    for (int q = 0; q < 4; ++q) {
      int idx = tid + q * 256; int r = idx >> 5, k = idx & 31;
      int kg = kk * 32 + k;
      float v = k1[(r0 + r) * 128 + kg];
      Abuf[r * 32 + k] = (bf16)fmaxf(ssc[kg] * v + ssh[kg], 0.f);
    }
    async_wait0();
    __syncthreads();
    v16bf a = load_a_frag(Abuf + wtR * 16 * 32, 32, lane);
    v16bf bb = load_b_frag_rm(Bbuf + wtC * 16 * 32, lane);
    acc = wmma_bf16(a, bb, acc);
    __syncthreads();
  }
  int m = lane & 15, half = lane >> 4;
#pragma unroll
  for (int v = 0; v < 8; ++v)
    k2out[(r0 + wtR * 16 + v + 8 * half) * 512 + oc0 + wtC * 16 + m] = (bf16)acc[v];
}

// Online per-segment-per-channel softmax stats over L rows (split over chunks)
__global__ void segsoft_partial(const bf16* __restrict__ k2, const float* sc_k2,
                                float* pm, float* ps, int L, int chunkRows) {
  int b = blockIdx.x, chunk = blockIdx.y, ch = threadIdx.x;  // 512 threads
  int r0 = chunk * chunkRows;
  float scale = sc_k2[ch], shift = sc_k2[512 + ch];
  const bf16* base = k2 + ((size_t)b * L + r0) * 512 + ch;
  float m = -1e30f, s = 0.f;
  for (int r = 0; r < chunkRows; ++r) {
    __builtin_prefetch((const void*)(base + (size_t)(r + 16) * 512), 0, 0);
    float v = fmaxf(scale * (float)base[(size_t)r * 512] + shift, 0.f);
    if (v > m) { s *= __expf(m - v); m = v; }
    s += __expf(v - m);
  }
  int o = (b * gridDim.y + chunk) * 512 + ch;
  pm[o] = m; ps[o] = s;
}

__global__ void segsoft_merge(const float* pm, const float* ps, float* mks, float* rsks,
                              const int* __restrict__ length, int nchunks) {
  int b = blockIdx.x, ch = threadIdx.x;
  float m = -1e30f;
  for (int k = 0; k < nchunks; ++k) m = fmaxf(m, pm[(b * nchunks + k) * 512 + ch]);
  float s = 0.f;
  for (int k = 0; k < nchunks; ++k)
    s += ps[(b * nchunks + k) * 512 + ch] * __expf(pm[(b * nchunks + k) * 512 + ch] - m);
  mks[b * 512 + ch] = m;
  rsks[b * 512 + ch] = 1.f / (s * (float)length[b]);
}

// Pooling einsum: pooled[b,ch,c] = (1/(s*L)) * sum_l exp(bnrelu(k2)-m) * flt[l,c]
// One block: 64 k-channels x 32 c, 8 waves (4 ch-tiles x 2 c-tiles), K-loop over L.
__global__ void einsum_kernel(const bf16* __restrict__ k2, const bf16* __restrict__ fltb,
                              const float* sc_k2, const float* mks, const float* rsks,
                              float* __restrict__ pooled, int L) {
  __shared__ __align__(16) bf16 Abuf[64 * 32];   // [ch][l] transposed tile
  __shared__ __align__(16) bf16 Bbuf[32 * 32];   // [l][c] (K x N, memory order)
  __shared__ float ssc[64], ssh[64], smk[64];
  int b = blockIdx.x, ch0 = blockIdx.y * 64;
  int tid = threadIdx.x, lane = tid & 31, wave = tid >> 5;
  int wtCh = wave >> 1, wtC = wave & 1;
  if (tid < 64) {
    ssc[tid] = sc_k2[ch0 + tid];
    ssh[tid] = sc_k2[512 + ch0 + tid];
    smk[tid] = mks[b * 512 + ch0 + tid];
  }
  __syncthreads();
  v8f acc = {};
  for (int l0 = 0; l0 < L; l0 += 32) {
    size_t rowbase = (size_t)b * L + l0;
#if HAS_ASYNC_LDS
    // B tile: 32 rows x 32 bf16 = 2KB contiguous; async while A does exp transform
    if (tid < 128) async_copy16(fltb + rowbase * 32 + (size_t)tid * 8, (char*)Bbuf + tid * 16);
#else
#pragma unroll
    for (int q = 0; q < 4; ++q) {
      int idx = tid + q * 256; int l = idx >> 5, cc = idx & 31;
      Bbuf[l * 32 + cc] = fltb[(rowbase + l) * 32 + cc];
    }
#endif
#pragma unroll
    for (int q = 0; q < 8; ++q) {
      int idx = tid + q * 256; int l = idx >> 6, j = idx & 63;
      float v = (float)k2[(rowbase + l) * 512 + ch0 + j];
      v = fmaxf(ssc[j] * v + ssh[j], 0.f);
      Abuf[j * 32 + l] = (bf16)__expf(v - smk[j]);
    }
    async_wait0();
    __syncthreads();
    v16bf a = load_a_frag(Abuf + wtCh * 16 * 32, 32, lane);
    v16bf bb = load_b_frag_kn(Bbuf + wtC * 16, 32, lane);
    acc = wmma_bf16(a, bb, acc);
    __syncthreads();
  }
  int m = lane & 15, half = lane >> 4;
#pragma unroll
  for (int v = 0; v < 8; ++v) {
    int ch = ch0 + wtCh * 16 + v + 8 * half;
    int cc = wtC * 16 + m;
    pooled[((size_t)b * 512 + ch) * 32 + cc] = acc[v] * rsks[b * 512 + ch];
  }
}

// GEMM3: r1_pre[16,512] = pooled_bf16[16,16384] @ fc_w1^T   (M=16, K=16384)
__global__ void gemm3_kernel(const bf16* __restrict__ A, const bf16* __restrict__ W,
                             float* __restrict__ r1) {
  __shared__ __align__(16) bf16 Abuf[16 * 32];
  __shared__ __align__(16) bf16 Bbuf[128 * 32];  // row-major [oc][k-chunk]
  int tid = threadIdx.x, lane = tid & 31, wave = tid >> 5;
  int oc0 = blockIdx.x * 128;
  v8f acc = {};
  for (int kk = 0; kk < 512; ++kk) {
    int kb = kk * 32;
#if HAS_ASYNC_LDS
    if (tid < 64) {
      int row = tid >> 2, c4 = tid & 3;
      async_copy16(A + (size_t)row * 16384 + kb + c4 * 8, (char*)Abuf + (row * 32 + c4 * 8) * 2);
    }
#pragma unroll
    for (int q = 0; q < 2; ++q) {
      int idx = tid + q * 256; int row = idx >> 2, c4 = idx & 3;
      async_copy16(W + (size_t)(oc0 + row) * 16384 + kb + c4 * 8,
                   (char*)Bbuf + (row * 32 + c4 * 8) * 2);
    }
    async_wait0();
#else
#pragma unroll
    for (int q = 0; q < 2; ++q) {
      int idx = tid + q * 256;
      if (idx < 512) { int r = idx >> 5, k = idx & 31; Abuf[r * 32 + k] = A[r * 16384 + kb + k]; }
    }
#pragma unroll
    for (int q = 0; q < 16; ++q) {
      int idx = tid + q * 256; int j = idx >> 5, k = idx & 31;
      Bbuf[j * 32 + k] = W[(size_t)(oc0 + j) * 16384 + kb + k];
    }
#endif
    __syncthreads();
    v16bf a = load_a_frag(Abuf, 32, lane);
    v16bf bb = load_b_frag_rm(Bbuf + wave * 16 * 32, lane);
    acc = wmma_bf16(a, bb, acc);
    __syncthreads();
  }
  int m = lane & 15, half = lane >> 4;
#pragma unroll
  for (int v = 0; v < 8; ++v)
    r1[(v + 8 * half) * 512 + oc0 + wave * 16 + m] = acc[v];
}

// Final: BN(r1)+relu -> @ fc_w2^T -> BN -> L2 normalize.  B fixed = 16.
__global__ void final_kernel(const float* __restrict__ r1_pre,
                             const float* fc_g1, const float* fc_be1,
                             const float* __restrict__ fc_w2,
                             const float* fc_g2, const float* fc_be2,
                             float* __restrict__ out) {
  __shared__ float sh1[16 * 512];
  __shared__ float sh2[16 * 256];
  __shared__ float rn[16];
  int tid = threadIdx.x;  // 256
  for (int c = tid; c < 512; c += 256) {
    float s = 0.f, q = 0.f;
    for (int r = 0; r < 16; ++r) { float v = r1_pre[r * 512 + c]; s += v; q += v * v; }
    float mu = s * (1.f / 16.f), var = q * (1.f / 16.f) - mu * mu;
    float sc = fc_g1[c] * rsqrtf(var + 1e-5f), shf = fc_be1[c] - mu * sc;
    for (int r = 0; r < 16; ++r) sh1[r * 512 + c] = fmaxf(r1_pre[r * 512 + c] * sc + shf, 0.f);
  }
  __syncthreads();
  {
    int oc = tid;
    for (int r = 0; r < 16; ++r) {
      float acc = 0.f;
      for (int k = 0; k < 512; ++k) acc += sh1[r * 512 + k] * fc_w2[oc * 512 + k];
      sh2[r * 256 + oc] = acc;
    }
  }
  __syncthreads();
  {
    int c = tid;
    float s = 0.f, q = 0.f;
    for (int r = 0; r < 16; ++r) { float v = sh2[r * 256 + c]; s += v; q += v * v; }
    float mu = s * (1.f / 16.f), var = q * (1.f / 16.f) - mu * mu;
    float sc = fc_g2[c] * rsqrtf(var + 1e-5f), shf = fc_be2[c] - mu * sc;
    for (int r = 0; r < 16; ++r) sh2[r * 256 + c] = sh2[r * 256 + c] * sc + shf;
  }
  __syncthreads();
  if (tid < 16) {
    float q = 0.f;
    for (int c = 0; c < 256; ++c) { float v = sh2[tid * 256 + c]; q += v * v; }
    rn[tid] = 1.f / fmaxf(sqrtf(q), 1e-12f);
  }
  __syncthreads();
  for (int r = 0; r < 16; ++r) out[r * 256 + tid] = sh2[r * 256 + tid] * rn[r];
}

// ---------------------------------------------------------------------------
extern "C" void kernel_launch(void* const* d_in, const int* in_sizes, int n_in,
                              void* d_out, int out_size, void* d_ws, size_t ws_size,
                              hipStream_t stream) {
  const float* x      = (const float*)d_in[0];
  const int*   length = (const int*)d_in[1];
  const float* po_w1  = (const float*)d_in[2];
  // d_in[3] po_b1 absorbed by BN
  const float* po_g1  = (const float*)d_in[4];
  const float* po_be1 = (const float*)d_in[5];
  const float* po_w2  = (const float*)d_in[6];
  const float* po_g2  = (const float*)d_in[8];
  const float* po_be2 = (const float*)d_in[9];
  const float* ch_w1  = (const float*)d_in[10];
  const float* ch_b1  = (const float*)d_in[11];
  const float* ch_w2  = (const float*)d_in[12];
  const float* ch_b2  = (const float*)d_in[13];
  const float* k_w1   = (const float*)d_in[14];
  const float* k_g1   = (const float*)d_in[16];
  const float* k_be1  = (const float*)d_in[17];
  const float* k_w2   = (const float*)d_in[18];
  const float* k_g2   = (const float*)d_in[20];
  const float* k_be2  = (const float*)d_in[21];
  const float* fc_w1  = (const float*)d_in[22];
  const float* fc_g1  = (const float*)d_in[24];
  const float* fc_be1 = (const float*)d_in[25];
  const float* fc_w2  = (const float*)d_in[26];
  const float* fc_g2  = (const float*)d_in[28];
  const float* fc_be2 = (const float*)d_in[29];

  const int N = in_sizes[0] / 32;   // 131072
  const int B = in_sizes[1];        // 16
  const int L = N / B;              // 8192

  // ---- workspace bump allocator ----
  char* base = (char*)d_ws;
  size_t off = 0;
  auto alloc = [&](size_t bytes) -> char* {
    off = (off + 255) & ~(size_t)255;
    char* p = base + off;
    off += bytes;
    return p;
  };

  size_t nz = 32 + 2 + 256 + 1024 + (size_t)B * 32;
  float* statz = (float*)alloc(nz * 4);
  float* s_po1 = statz;
  float* s_po2 = statz + 32;
  float* s_k1  = statz + 34;
  float* s_k2  = statz + 34 + 256;
  float* csum  = statz + 34 + 256 + 1024;

  float* sc_po1 = (float*)alloc(32 * 4);
  float* sc_po2 = (float*)alloc(2 * 4);
  float* sc_k1  = (float*)alloc(256 * 4);
  float* sc_k2  = (float*)alloc(1024 * 4);
  float* c_att  = (float*)alloc((size_t)B * 32 * 4);
  float* h1     = (float*)alloc((size_t)N * 16 * 4);
  float* zb     = (float*)alloc((size_t)N * 4);
  float* p_soft = (float*)alloc((size_t)N * 4);
  bf16*  fltb   = (bf16*)alloc((size_t)N * 32 * 2);
  float* k1p    = (float*)alloc((size_t)N * 128 * 4);
  bf16*  k2b    = (bf16*)alloc((size_t)N * 512 * 2);
  float* pm     = (float*)alloc((size_t)B * 8 * 512 * 4);
  float* ps     = (float*)alloc((size_t)B * 8 * 512 * 4);
  float* mks    = (float*)alloc((size_t)B * 512 * 4);
  float* rsks   = (float*)alloc((size_t)B * 512 * 4);
  float* pooled = (float*)alloc((size_t)B * 512 * 32 * 4);
  bf16*  pooledb= (bf16*)alloc((size_t)B * 16384 * 2);
  float* r1p    = (float*)alloc((size_t)B * 512 * 4);
  bf16*  w1b    = (bf16*)alloc((size_t)128 * 32 * 2);
  bf16*  w2b    = (bf16*)alloc((size_t)512 * 128 * 2);
  bf16*  fw1b   = (bf16*)alloc((size_t)512 * 16384 * 2);
  (void)ws_size; (void)n_in; (void)out_size;

  // ---- pipeline ----
  zero_kernel<<<(int)((nz + 255) / 256), 256, 0, stream>>>(statz, (int)nz);
  f2bf_kernel<<<16, 256, 0, stream>>>(k_w1, w1b, 128 * 32);
  f2bf_kernel<<<256, 256, 0, stream>>>(k_w2, w2b, 512 * 128);
  f2bf_kernel<<<4096, 256, 0, stream>>>(fc_w1, fw1b, (long long)512 * 16384);

  po1_kernel<<<N / 256, 256, 0, stream>>>(x, po_w1, h1, s_po1, csum, L);
  bn_finalize<<<1, 32, 0, stream>>>(s_po1, po_g1, po_be1, sc_po1, 16, 1.0f / N);
  po2_kernel<<<N / 256, 256, 0, stream>>>(h1, sc_po1, po_w2, zb, s_po2);
  bn_finalize<<<1, 32, 0, stream>>>(s_po2, po_g2, po_be2, sc_po2, 1, 1.0f / N);
  pseg_kernel<<<B, 256, 0, stream>>>(zb, sc_po2, p_soft, L);
  chmlp_kernel<<<1, 32, 0, stream>>>(csum, ch_w1, ch_b1, ch_w2, ch_b2, c_att, B, 1.0f / L);
  flt_kernel<<<(N * 32) / 256, 256, 0, stream>>>(x, p_soft, c_att, fltb, L);

  gemm1_kernel<<<dim3(N / 32, 2), 256, 0, stream>>>(fltb, w1b, k1p);
  stats_f32_kernel<<<1024, 256, 0, stream>>>(k1p, s_k1, (long long)N * 128, 128);
  bn_finalize<<<1, 128, 0, stream>>>(s_k1, k_g1, k_be1, sc_k1, 128, 1.0f / N);

  gemm2_kernel<<<dim3(N / 32, 8), 256, 0, stream>>>(k1p, w2b, sc_k1, k2b);
  stats_bf16_kernel<<<2048, 256, 0, stream>>>(k2b, s_k2, (long long)N * 512, 512);
  bn_finalize<<<1, 512, 0, stream>>>(s_k2, k_g2, k_be2, sc_k2, 512, 1.0f / N);

  segsoft_partial<<<dim3(B, 8), 512, 0, stream>>>(k2b, sc_k2, pm, ps, L, L / 8);
  segsoft_merge<<<B, 512, 0, stream>>>(pm, ps, mks, rsks, length, 8);

  einsum_kernel<<<dim3(B, 8), 256, 0, stream>>>(k2b, fltb, sc_k2, mks, rsks, pooled, L);
  f2bf_kernel<<<(B * 16384) / 256, 256, 0, stream>>>(pooled, pooledb, (long long)B * 16384);

  gemm3_kernel<<<4, 256, 0, stream>>>(pooledb, fw1b, r1p);
  final_kernel<<<1, 256, 0, stream>>>(r1p, fc_g1, fc_be1, fc_w2, fc_g2, fc_be2, (float*)d_out);
}
